// LSTMCell_12790412608061
// MI455X (gfx1250) — matrix operand order repaired and checked by reading, not compiled, not run on hardware
//
#include <hip/hip_runtime.h>
#include <stdint.h>

// ---------------------------------------------------------------------------
// Quantized LSTM cell for MI455X (gfx1250, wave32, WMMA).
// B*W = 524288 rows, K = I+H = 128, N = 4 gates * 64 = 256.
// Memory-bound (~640MiB mandatory traffic @23.3TB/s ~ 29us); f16 WMMA GEMM
// recomputed twice (minmax pass + emit pass) instead of materializing 512MiB
// of raw gates. Monotone observers folded analytically on one thread.
// ---------------------------------------------------------------------------

typedef __attribute__((ext_vector_type(16))) _Float16 v16h;
typedef __attribute__((ext_vector_type(8)))  _Float16 v8h;
typedef __attribute__((ext_vector_type(8)))  float    v8f;

// qparam slots: per slot {minKey(u32), maxKey(u32), scale(f32bits), zp(f32bits)}
#define QP_X   0
#define QP_H   1
#define QP_C   2
#define QP_W0  3   // W_i,W_f,W_o,W_c = 3..6
#define QP_B0  7   // b_i,b_f,b_o,b_c = 7..10
#define QP_CAT 11
#define QP_G0  12  // gate preact i,f,o,c = 12..15
#define QP_A0  16  // activations i,f,o,c = 16..19
#define QP_FC  20
#define QP_IC  21
#define QP_CN  22
#define QP_CT  23
#define QP_HN  24
#define NSLOT  25

// ---- ordered-uint encoding of floats for atomicMin/Max --------------------
__device__ inline uint32_t fkey(float f) {
    uint32_t u = __float_as_uint(f);
    return (u & 0x80000000u) ? ~u : (u | 0x80000000u);
}
__device__ inline float funkey(uint32_t k) {
    uint32_t u = (k & 0x80000000u) ? (k & 0x7FFFFFFFu) : ~k;
    return __uint_as_float(u);
}

__device__ inline float qscale(const uint32_t* qp, int s) { return __uint_as_float(qp[s * 4 + 2]); }
__device__ inline float qzp(const uint32_t* qp, int s)    { return __uint_as_float(qp[s * 4 + 3]); }

// fake_quant: q = clip(rint(x/s)+zp, -128, 127); out = (q-zp)*s
__device__ inline float fq_apply(float x, float s, float zp) {
    float q = fminf(fmaxf(rintf(x / s) + zp, -128.f), 127.f);
    return (q - zp) * s;
}
__device__ inline float fq_code(float x, float s, float zp) {
    return fminf(fmaxf(rintf(x / s) + zp, -128.f), 127.f);
}
__device__ inline float hsig(float x)  { return fminf(fmaxf(x / 6.f + 0.5f, 0.f), 1.f); }
__device__ inline float htanh(float x) { return fminf(fmaxf(x, -1.f), 1.f); }

// wave32 min/max reduce then one atomic per wave
__device__ inline void wave_minmax_push(float mn, float mx, uint32_t* qp, int slot) {
#pragma unroll
    for (int off = 16; off; off >>= 1) {
        mn = fminf(mn, __shfl_xor(mn, off, 32));
        mx = fmaxf(mx, __shfl_xor(mx, off, 32));
    }
    if ((threadIdx.x & 31u) == 0) {
        atomicMin(&qp[slot * 4 + 0], fkey(mn));
        atomicMax(&qp[slot * 4 + 1], fkey(mx));
    }
}

__device__ inline void set_sz(uint32_t* qp, int slot, float mn, float mx) {
    mn = fminf(mn, 0.f);
    mx = fmaxf(mx, 0.f);
    float s = fmaxf((mx - mn) / 255.f, 1e-8f);
    float zp = fminf(fmaxf(rintf(-128.f - mn / s), -128.f), 127.f);
    qp[slot * 4 + 2] = __float_as_uint(s);
    qp[slot * 4 + 3] = __float_as_uint(zp);
}
__device__ inline void set_from_keys(uint32_t* qp, int slot) {
    set_sz(qp, slot, funkey(qp[slot * 4 + 0]), funkey(qp[slot * 4 + 1]));
}

// ---------------------------------------------------------------------------
__global__ void k_init(uint32_t* qp) {
    int i = threadIdx.x;
    if (i < NSLOT) {
        qp[i * 4 + 0] = 0xFFFFFFFFu;      // minKey = +inf side
        qp[i * 4 + 1] = 0u;               // maxKey = -inf side
        qp[i * 4 + 2] = __float_as_uint(1.f);
        qp[i * 4 + 3] = 0u;
    }
}

__global__ void k_minmax(const float* __restrict__ x, long long n, uint32_t* qp, int slot) {
    float mn = 1e30f, mx = -1e30f;
    long long stride = (long long)gridDim.x * blockDim.x;
    for (long long i = (long long)blockIdx.x * blockDim.x + threadIdx.x; i < n; i += stride) {
        float v = x[i];
        mn = fminf(mn, v);
        mx = fmaxf(mx, v);
    }
    wave_minmax_push(mn, mx, qp, slot);
}

// monotone-observer folding on one thread
__global__ void k_finalize(uint32_t* qp, int stage) {
    if (threadIdx.x != 0 || blockIdx.x != 0) return;
    if (stage == 0) {
        for (int s = 0; s <= 10; ++s) set_from_keys(qp, s);
        float xs = qscale(qp, QP_X), xz = qzp(qp, QP_X);
        float hs = qscale(qp, QP_H), hz = qzp(qp, QP_H);
        float xmn = funkey(qp[QP_X * 4]), xmx = funkey(qp[QP_X * 4 + 1]);
        float hmn = funkey(qp[QP_H * 4]), hmx = funkey(qp[QP_H * 4 + 1]);
        // cat = fq(concat(fq(x),fq(h))): fq is monotone -> derive extremes
        float cmn = fminf(fq_apply(xmn, xs, xz), fq_apply(hmn, hs, hz));
        float cmx = fmaxf(fq_apply(xmx, xs, xz), fq_apply(hmx, hs, hz));
        set_sz(qp, QP_CAT, cmn, cmx);
    } else if (stage == 1) {
        for (int g = 0; g < 4; ++g) {
            set_from_keys(qp, QP_G0 + g);
            float s = qscale(qp, QP_G0 + g), z = qzp(qp, QP_G0 + g);
            float gmn = fq_apply(funkey(qp[(QP_G0 + g) * 4 + 0]), s, z);
            float gmx = fq_apply(funkey(qp[(QP_G0 + g) * 4 + 1]), s, z);
            float amn = (g == 3) ? htanh(gmn) : hsig(gmn);
            float amx = (g == 3) ? htanh(gmx) : hsig(gmx);
            set_sz(qp, QP_A0 + g, amn, amx);
        }
    } else if (stage == 2) {
        set_from_keys(qp, QP_FC);
        set_from_keys(qp, QP_IC);
    } else if (stage == 3) {
        set_from_keys(qp, QP_CN);
        float s = qscale(qp, QP_CN), z = qzp(qp, QP_CN);
        float tmn = htanh(fq_apply(funkey(qp[QP_CN * 4 + 0]), s, z));
        float tmx = htanh(fq_apply(funkey(qp[QP_CN * 4 + 1]), s, z));
        set_sz(qp, QP_CT, tmn, tmx);
    } else {
        set_from_keys(qp, QP_HN);
    }
}

// build f16 cat[rows][128] with fq(x)/fq(h)+fq(cat) folded in
__global__ void k_pack_cat(const float* __restrict__ x, const float* __restrict__ h,
                           _Float16* __restrict__ cat, const uint32_t* __restrict__ qp,
                           long long rows) {
    float xs = qscale(qp, QP_X), xz = qzp(qp, QP_X);
    float hs = qscale(qp, QP_H), hz = qzp(qp, QP_H);
    float cs = qscale(qp, QP_CAT), cz = qzp(qp, QP_CAT);
    long long n = rows * 128;
    long long stride = (long long)gridDim.x * blockDim.x;
    for (long long i = (long long)blockIdx.x * blockDim.x + threadIdx.x; i < n; i += stride) {
        long long row = i >> 7;
        int k = (int)(i & 127);
        float v = (k < 64) ? fq_apply(x[row * 64 + k], xs, xz)
                           : fq_apply(h[row * 64 + (k - 64)], hs, hz);
        cat[i] = (_Float16)fq_apply(v, cs, cz);
    }
}

// pack fq(W) into exact WMMA B-lane order: [ks 0..3][nt 0..15][lane 0..31][e 0..15]
// element (K=ks*32+table-order k(lane/16,e), N=nt*16+lane%16) ; plus fq(bias)
__global__ void k_pack_w(const float* Wi, const float* Wf, const float* Wo, const float* Wc,
                         const float* bi, const float* bf, const float* bo, const float* bc,
                         _Float16* __restrict__ wp, float* __restrict__ bp,
                         const uint32_t* __restrict__ qp) {
    int idx = blockIdx.x * blockDim.x + threadIdx.x;
    if (idx < 128 * 256) {
        int N = idx >> 7, K = idx & 127;
        int gate = N >> 6, hh = N & 63;
        const float* W = gate == 0 ? Wi : gate == 1 ? Wf : gate == 2 ? Wo : Wc;
        float v = fq_apply(W[hh * 128 + K], qscale(qp, QP_W0 + gate), qzp(qp, QP_W0 + gate));
        int ks = K >> 5, kk = K & 31, nt = N >> 4;
        int lane = (N & 15) + ((kk >> 3) & 1) * 16, e = (kk & 7) + (kk >> 4) * 8;
        wp[(((ks * 16) + nt) * 32 + lane) * 16 + e] = (_Float16)v;
    }
    if (idx < 256) {
        int gate = idx >> 6;
        const float* b = gate == 0 ? bi : gate == 1 ? bf : gate == 2 ? bo : bc;
        bp[idx] = fq_apply(b[idx & 63], qscale(qp, QP_B0 + gate), qzp(qp, QP_B0 + gate));
    }
}

// ---------------------------------------------------------------------------
// WMMA GEMM: 8 waves/block, each wave a 16-row tile; K=128 in 4 x (16x16x32).
// Outer loop over gates (4 x 16 cols), inner fully-unrolled loop over the 4
// 16-col tiles of that gate so the compiler can interleave the 16 ds_load_b128
// pairs with the 16 chained WMMAs (hides LDS latency), and min/max reduction
// runs once per gate instead of once per tile.
// PHASE 0: reduce per-gate preact min/max only (no stores).
// PHASE 1: fq(preact)->activation->fq->int8 code, stored to d_out h-half
//          (256 B/row = 4 gates x 64 codes); o-gate codes duplicated to ws.
// ---------------------------------------------------------------------------
template <int PHASE>
__global__ __launch_bounds__(256) void k_gemm(const _Float16* __restrict__ cat,
                                              const _Float16* __restrict__ wp,
                                              const float* __restrict__ bp,
                                              uint32_t* qp,
                                              uint8_t* __restrict__ outCodes,
                                              uint8_t* __restrict__ oCodes,
                                              long long rows) {
    __shared__ __align__(32) _Float16 Bs[128 * 256];   // 64KB B panel in LDS
    {   // cooperative 64KB block copy (already in final lane order)
        const uint4* src = (const uint4*)wp;
        uint4* dst = (uint4*)Bs;
        for (int i = threadIdx.x; i < 4096; i += 256) dst[i] = src[i];
    }
    __syncthreads();

    int wave = threadIdx.x >> 5;
    int lane = threadIdx.x & 31;
    int half = lane >> 4;      // lanes 0-15 vs 16-31
    int mrow = lane & 15;
    long long rowBase = ((long long)blockIdx.x * 8 + wave) * 16;

    // A: 16x32 f16 per kstep. ISA layout: lanes0-15 e0..7 = K0..7, e8..15 = K16..23;
    // lanes16-31 e0..7 = K8..15, e8..15 = K24..31  => two 16B loads per lane.
    v16h A[4];
    const _Float16* arow = cat + (rowBase + mrow) * 128;
    __builtin_prefetch(arow + 8 * 16 * 128, 0, 1);   // speculative next block tile
#pragma unroll
    for (int ks = 0; ks < 4; ++ks) {
        v8h lo = *(const v8h*)(arow + ks * 32 + half * 8);
        v8h hi = *(const v8h*)(arow + ks * 32 + half * 8 + 16);
        A[ks] = __builtin_shufflevector(lo, hi, 0, 1, 2, 3, 4, 5, 6, 7,
                                        8, 9, 10, 11, 12, 13, 14, 15);
    }

#pragma unroll 1
    for (int gate = 0; gate < 4; ++gate) {
        float gsc = 0.f, gzp = 0.f, asc = 0.f, azp = 0.f;
        if (PHASE == 1) {            // scalar loads per gate (L2-hot, avoids
            gsc = qscale(qp, QP_G0 + gate);   // runtime-indexed register arrays)
            gzp = qzp(qp, QP_G0 + gate);
            asc = qscale(qp, QP_A0 + gate);
            azp = qzp(qp, QP_A0 + gate);
        }
        float mn = 1e30f, mx = -1e30f;
#pragma unroll
        for (int t = 0; t < 4; ++t) {
            const int nt = gate * 4 + t;
            v8f acc = {};
#pragma unroll
            for (int ks = 0; ks < 4; ++ks) {
                v16h B = *(const v16h*)&Bs[(((ks * 16) + nt) * 32 + lane) * 16];
                acc = __builtin_amdgcn_wmma_f32_16x16x32_f16(
                    false, A[ks], false, B, (short)0, acc, false, false);
            }
            int col = nt * 16 + mrow;        // C/D layout: N = lane&15
            float bias = bp[col];
            if (PHASE == 0) {
#pragma unroll
                for (int v = 0; v < 8; ++v) {   // M = v + 8*half
                    float val = acc[v] + bias;
                    mn = fminf(mn, val);
                    mx = fmaxf(mx, val);
                }
            } else {
                int cg = col & 63;
#pragma unroll
                for (int v = 0; v < 8; ++v) {
                    long long row = rowBase + half * 8 + v;
                    float val = acc[v] + bias;
                    float gv = fq_apply(val, gsc, gzp);
                    float a = (gate == 3) ? htanh(gv) : hsig(gv);
                    uint8_t code = (uint8_t)(int)fq_code(a, asc, azp);
                    outCodes[row * 256 + gate * 64 + cg] = code;
                    if (gate == 2) oCodes[row * 64 + cg] = code;
                }
            }
        }
        if (PHASE == 0) wave_minmax_push(mn, mx, qp, QP_G0 + gate);
    }
}

// ---- elementwise passes over int8 codes (cheap: ~160MiB reads each) -------
__global__ void k_fcic(const uint8_t* __restrict__ codes, const float* __restrict__ cprev,
                       uint32_t* qp, long long n) {
    float cs = qscale(qp, QP_C), cz = qzp(qp, QP_C);
    float is_ = qscale(qp, QP_A0 + 0), iz = qzp(qp, QP_A0 + 0);
    float fs = qscale(qp, QP_A0 + 1), fz = qzp(qp, QP_A0 + 1);
    float gs = qscale(qp, QP_A0 + 3), gz = qzp(qp, QP_A0 + 3);
    float fcmn = 1e30f, fcmx = -1e30f, icmn = 1e30f, icmx = -1e30f;
    long long stride = (long long)gridDim.x * blockDim.x;
    for (long long i = (long long)blockIdx.x * blockDim.x + threadIdx.x; i < n; i += stride) {
        long long row = i >> 6; int j = (int)(i & 63);
        const uint8_t* rc = codes + row * 256;
        float iv = ((float)(int8_t)rc[j] - iz) * is_;
        float fv = ((float)(int8_t)rc[64 + j] - fz) * fs;
        float gv = ((float)(int8_t)rc[192 + j] - gz) * gs;
        float cv = fq_apply(cprev[i], cs, cz);
        float fc = fv * cv, ic = iv * gv;
        fcmn = fminf(fcmn, fc); fcmx = fmaxf(fcmx, fc);
        icmn = fminf(icmn, ic); icmx = fmaxf(icmx, ic);
    }
    wave_minmax_push(fcmn, fcmx, qp, QP_FC);
    wave_minmax_push(icmn, icmx, qp, QP_IC);
}

__global__ void k_cnext_mm(const uint8_t* __restrict__ codes, const float* __restrict__ cprev,
                           uint32_t* qp, long long n) {
    float cs = qscale(qp, QP_C), cz = qzp(qp, QP_C);
    float is_ = qscale(qp, QP_A0 + 0), iz = qzp(qp, QP_A0 + 0);
    float fs = qscale(qp, QP_A0 + 1), fz = qzp(qp, QP_A0 + 1);
    float gs = qscale(qp, QP_A0 + 3), gz = qzp(qp, QP_A0 + 3);
    float fcs = qscale(qp, QP_FC), fcz = qzp(qp, QP_FC);
    float ics = qscale(qp, QP_IC), icz = qzp(qp, QP_IC);
    float mn = 1e30f, mx = -1e30f;
    long long stride = (long long)gridDim.x * blockDim.x;
    for (long long i = (long long)blockIdx.x * blockDim.x + threadIdx.x; i < n; i += stride) {
        long long row = i >> 6; int j = (int)(i & 63);
        const uint8_t* rc = codes + row * 256;
        float iv = ((float)(int8_t)rc[j] - iz) * is_;
        float fv = ((float)(int8_t)rc[64 + j] - fz) * fs;
        float gv = ((float)(int8_t)rc[192 + j] - gz) * gs;
        float cv = fq_apply(cprev[i], cs, cz);
        float cn = fq_apply(fv * cv, fcs, fcz) + fq_apply(iv * gv, ics, icz);
        mn = fminf(mn, cn); mx = fmaxf(mx, cn);
    }
    wave_minmax_push(mn, mx, qp, QP_CN);
}

__global__ void k_cnext_store(const uint8_t* __restrict__ codes, const float* __restrict__ cprev,
                              const uint8_t* __restrict__ oc, float* __restrict__ out_c,
                              uint32_t* qp, long long n) {
    float cs = qscale(qp, QP_C), cz = qzp(qp, QP_C);
    float is_ = qscale(qp, QP_A0 + 0), iz = qzp(qp, QP_A0 + 0);
    float fs = qscale(qp, QP_A0 + 1), fz = qzp(qp, QP_A0 + 1);
    float gs = qscale(qp, QP_A0 + 3), gz = qzp(qp, QP_A0 + 3);
    float os_ = qscale(qp, QP_A0 + 2), oz = qzp(qp, QP_A0 + 2);
    float fcs = qscale(qp, QP_FC), fcz = qzp(qp, QP_FC);
    float ics = qscale(qp, QP_IC), icz = qzp(qp, QP_IC);
    float cns = qscale(qp, QP_CN), cnz = qzp(qp, QP_CN);
    float cts = qscale(qp, QP_CT), ctz = qzp(qp, QP_CT);
    float mn = 1e30f, mx = -1e30f;
    long long stride = (long long)gridDim.x * blockDim.x;
    for (long long i = (long long)blockIdx.x * blockDim.x + threadIdx.x; i < n; i += stride) {
        long long row = i >> 6; int j = (int)(i & 63);
        const uint8_t* rc = codes + row * 256;
        float iv = ((float)(int8_t)rc[j] - iz) * is_;
        float fv = ((float)(int8_t)rc[64 + j] - fz) * fs;
        float gv = ((float)(int8_t)rc[192 + j] - gz) * gs;
        float cv = fq_apply(cprev[i], cs, cz);
        float cn = fq_apply(fq_apply(fv * cv, fcs, fcz) + fq_apply(iv * gv, ics, icz), cns, cnz);
        out_c[i] = cn;
        float t = fq_apply(htanh(cn), cts, ctz);
        float ov = ((float)(int8_t)oc[i] - oz) * os_;
        float hr = ov * t;
        mn = fminf(mn, hr); mx = fmaxf(mx, hr);
    }
    wave_minmax_push(mn, mx, qp, QP_HN);
}

__global__ void k_h_store(const uint8_t* __restrict__ oc, const float* __restrict__ out_c,
                          float* __restrict__ out_h, const uint32_t* __restrict__ qp,
                          long long n) {
    float os_ = qscale(qp, QP_A0 + 2), oz = qzp(qp, QP_A0 + 2);
    float cts = qscale(qp, QP_CT), ctz = qzp(qp, QP_CT);
    float hs = qscale(qp, QP_HN), hz = qzp(qp, QP_HN);
    long long stride = (long long)gridDim.x * blockDim.x;
    for (long long i = (long long)blockIdx.x * blockDim.x + threadIdx.x; i < n; i += stride) {
        float t = fq_apply(htanh(out_c[i]), cts, ctz);
        float ov = ((float)(int8_t)oc[i] - oz) * os_;
        out_h[i] = fq_apply(ov * t, hs, hz);   // overwrites code area; oc lives in ws
    }
}

// ---------------------------------------------------------------------------
extern "C" void kernel_launch(void* const* d_in, const int* in_sizes, int n_in,
                              void* d_out, int out_size, void* d_ws, size_t ws_size,
                              hipStream_t stream) {
    const float* x  = (const float*)d_in[0];
    const float* hp = (const float*)d_in[1];
    const float* cp = (const float*)d_in[2];
    const float* Wg[4] = {(const float*)d_in[3], (const float*)d_in[5],
                          (const float*)d_in[7], (const float*)d_in[9]};
    const float* bg[4] = {(const float*)d_in[4], (const float*)d_in[6],
                          (const float*)d_in[8], (const float*)d_in[10]};

    long long rows = (long long)in_sizes[0] / 64;   // B*W (I = 64)
    long long n64 = rows * 64;

    // workspace layout (needs ~168MB): qparams | Wpack(64K) | bias(1K) | oCodes | catF16
    uint32_t* qp = (uint32_t*)d_ws;
    char* base = (char*)d_ws;
    _Float16* wp = (_Float16*)(base + 1024);
    float* bp = (float*)(base + 1024 + 65536);
    uint8_t* oc = (uint8_t*)(base + 1024 + 65536 + 1024);
    _Float16* cat = (_Float16*)(base + 67584 + (size_t)n64);

    uint8_t* codes = (uint8_t*)d_out;              // h-half as int8 gate codes (256B/row)
    float* out_h = (float*)d_out;
    float* out_c = (float*)d_out + n64;

    const int TB = 256;
    int gRed = (int)((n64 + TB - 1) / TB); if (gRed > 16384) gRed = 16384;
    int gCat = (int)((rows * 128 + TB - 1) / TB); if (gCat > 32768) gCat = 32768;
    int gGemm = (int)(rows / 128);                 // 16 rows/wave * 8 waves/block

    k_init<<<1, 32, 0, stream>>>(qp);
    k_minmax<<<gRed, TB, 0, stream>>>(x, n64, qp, QP_X);
    k_minmax<<<gRed, TB, 0, stream>>>(hp, n64, qp, QP_H);
    k_minmax<<<gRed, TB, 0, stream>>>(cp, n64, qp, QP_C);
    for (int g = 0; g < 4; ++g) {
        k_minmax<<<8, TB, 0, stream>>>(Wg[g], 64 * 128, qp, QP_W0 + g);
        k_minmax<<<1, 64, 0, stream>>>(bg[g], 64, qp, QP_B0 + g);
    }
    k_finalize<<<1, 1, 0, stream>>>(qp, 0);
    k_pack_cat<<<gCat, TB, 0, stream>>>(x, hp, cat, qp, rows);
    k_pack_w<<<128, TB, 0, stream>>>(Wg[0], Wg[1], Wg[2], Wg[3],
                                     bg[0], bg[1], bg[2], bg[3], wp, bp, qp);
    k_gemm<0><<<gGemm, TB, 0, stream>>>(cat, wp, bp, qp, codes, oc, rows);
    k_finalize<<<1, 1, 0, stream>>>(qp, 1);
    k_gemm<1><<<gGemm, TB, 0, stream>>>(cat, wp, bp, qp, codes, oc, rows);
    k_fcic<<<gRed, TB, 0, stream>>>(codes, cp, qp, n64);
    k_finalize<<<1, 1, 0, stream>>>(qp, 2);
    k_cnext_mm<<<gRed, TB, 0, stream>>>(codes, cp, qp, n64);
    k_finalize<<<1, 1, 0, stream>>>(qp, 3);
    k_cnext_store<<<gRed, TB, 0, stream>>>(codes, cp, oc, out_c, qp, n64);
    k_finalize<<<1, 1, 0, stream>>>(qp, 4);
    k_h_store<<<gRed, TB, 0, stream>>>(oc, out_c, out_h, qp, n64);
}
